// PointMessagePassing_50955492000255
// MI455X (gfx1250) — compile-verified, matrix operand order checked by reading, not run on hardware
//
#include <hip/hip_runtime.h>
#include <hip/hip_bf16.h>

typedef __attribute__((ext_vector_type(16))) _Float16 v16h;
typedef __attribute__((ext_vector_type(8)))  _Float16 v8h;
typedef __attribute__((ext_vector_type(4)))  _Float16 v4h;
typedef __attribute__((ext_vector_type(8)))  float    v8f;
typedef __attribute__((ext_vector_type(4)))  int      v4i;

#define DEMB   128
#define K1     256
#define MIDC   512
#define OUTC   128
#define NTYPES 8
#define TILE_M 128
#define A_STRIDE 264   // halves per A-tile row (256 + pad), 528B rows => 16B aligned
#define H_STRIDE 72    // halves per H-chunk row (64 + pad), 144B rows => 16B aligned

// gfx1250 async global->LDS path (ASYNCcnt); degrade to sync copy if builtins absent
#if defined(__HIP_DEVICE_COMPILE__) && \
    __has_builtin(__builtin_amdgcn_global_load_async_to_lds_b128) && \
    __has_builtin(__builtin_amdgcn_s_wait_asynccnt)
#define USE_ASYNC_LDS 1
typedef __attribute__((address_space(1))) v4i v4i_as1;   // global
typedef __attribute__((address_space(3))) v4i v4i_as3;   // LDS
#endif

union V16U { v16h v; v8h h[2]; };

// ---------------------------------------------------------------------------
// binning kernels
// ---------------------------------------------------------------------------
__global__ void zero_kernel(int* p) {
    if (threadIdx.x < 32) p[threadIdx.x] = 0;
}

__global__ void hist_kernel(const int* __restrict__ types, int nE, int* __restrict__ counts) {
    int i = blockIdx.x * blockDim.x + threadIdx.x;
    if (i < nE) atomicAdd(&counts[types[i] & 7], 1);
}

__global__ void prefix_kernel(const int* __restrict__ counts, int* __restrict__ offsets) {
    if (threadIdx.x == 0) {
        int s = 0;
        for (int i = 0; i < NTYPES; ++i) { offsets[i] = s; s += counts[i]; }
    }
}

__global__ void scatter_kernel(const int* __restrict__ types, int nE,
                               const int* __restrict__ offsets,
                               int* __restrict__ cursors, int* __restrict__ perm) {
    int i = blockIdx.x * blockDim.x + threadIdx.x;
    if (i < nE) {
        int t = types[i] & 7;
        int p = atomicAdd(&cursors[t], 1);
        perm[offsets[t] + p] = i;
    }
}

// ---------------------------------------------------------------------------
// weight packing: f32 row-major -> f16 WMMA B-fragment order.
// B fragment (32x16, 16-bit): lane l holds column n = l&15; slot s holds
// k = 16*(s>>3) + 8*(l>>4) + (s&7)   (mirrors the ISA 16-bit operand layout).
// Fragment f stored contiguously: w[f*512 + l*16 + s].
// ---------------------------------------------------------------------------
__device__ __forceinline__ int kmap(int l, int s) {
    return ((s >> 3) << 4) + ((l >> 4) << 3) + (s & 7);
}

// W1 frag order: frag = nt*8 + kt  (nt: MID/16, kt: K1/32)
// => each 64-col MID chunk is a contiguous 32KB region (frags chunk*32..+31).
__global__ void pack_w1_kernel(const float* __restrict__ W1, _Float16* __restrict__ w1p) {
    int i = blockIdx.x * blockDim.x + threadIdx.x;      // 256*512 halves total
    if (i >= K1 * MIDC) return;
    int s = i & 15, l = (i >> 4) & 31, frag = i >> 9;
    int kt = frag & 7, nt = frag >> 3;
    int k = kt * 32 + kmap(l, s);
    int n = nt * 16 + (l & 15);
    w1p[i] = (_Float16)W1[k * MIDC + n];
}

// W2 frag order: frag = ty*128 + ktg*8 + nn   (ty: 8, ktg: MID/32, nn: OUT/16)
// => each (type, 64-col MID chunk) group is a contiguous 16KB region.
__global__ void pack_w2_kernel(const float* __restrict__ W2, _Float16* __restrict__ w2p) {
    int i = blockIdx.x * blockDim.x + threadIdx.x;      // 512*1024 halves total
    if (i >= MIDC * OUTC * NTYPES) return;
    int s = i & 15, l = (i >> 4) & 31, frag = i >> 9;
    int nn  = frag & 7;
    int ktg = (frag >> 3) & 15;
    int ty  = frag >> 7;
    int k = ktg * 32 + kmap(l, s);
    int n = ty * OUTC + nn * 16 + (l & 15);
    w2p[i] = (_Float16)W2[k * (OUTC * NTYPES) + n];
}

// ---------------------------------------------------------------------------
// fragment loaders
// ---------------------------------------------------------------------------
__device__ __forceinline__ v16h load_afrag_lds(const _Float16* base, int row0, int k0,
                                               int lane, int stride) {
    int m = row0 + (lane & 15);
    int koff = k0 + ((lane >> 4) << 3);
    const _Float16* p = base + m * stride + koff;
    V16U u;
    u.h[0] = *(const v8h*)(p);
    u.h[1] = *(const v8h*)(p + 16);
    return u.v;
}

__device__ __forceinline__ v16h load_bfrag_lds(const _Float16* base, int frag, int lane) {
    return *(const v16h*)(base + (frag << 9) + lane * 16);
}

__device__ __forceinline__ void async_b128(const _Float16* src, _Float16* dst) {
#ifdef USE_ASYNC_LDS
    __builtin_amdgcn_global_load_async_to_lds_b128(
        (v4i_as1*)src, (v4i_as3*)dst, 0, 0);
#else
    *(v8h*)dst = *(const v8h*)src;
#endif
}

// stage one chunk's weight fragments (32KB W1 + 16KB W2) global -> LDS, once per block
__device__ __forceinline__ void stage_weights(const _Float16* __restrict__ gW1,
                                              const _Float16* __restrict__ gW2,
                                              _Float16* sW1, _Float16* sW2, int tid) {
#pragma unroll
    for (int j = 0; j < 8; ++j) {                 // 2048 x b128 for W1 chunk
        int idx = tid + j * 256;
        async_b128(gW1 + idx * 8, sW1 + idx * 8);
    }
#pragma unroll
    for (int j = 0; j < 4; ++j) {                 // 1024 x b128 for W2 chunk
        int idx = tid + j * 256;
        async_b128(gW2 + idx * 8, sW2 + idx * 8);
    }
#ifdef USE_ASYNC_LDS
    __builtin_amdgcn_s_wait_asynccnt(0);
#endif
}

// ---------------------------------------------------------------------------
// fused MLP over one 128-row, single-edge-type tile
// ---------------------------------------------------------------------------
__global__ __launch_bounds__(256) void mlp_tile_kernel(
    const float* __restrict__ from_e, const float* __restrict__ to_e,
    const int* __restrict__ perm, const int* __restrict__ counts,
    const int* __restrict__ offsets,
    const _Float16* __restrict__ w1p, const _Float16* __restrict__ w2p,
    const float* __restrict__ b1, const float* __restrict__ b2,
    float* __restrict__ out) {

    __shared__ _Float16 Alds[TILE_M * A_STRIDE];   // 66 KB, f16 concat(from,to)
    __shared__ _Float16 Hlds[TILE_M * H_STRIDE];   // 18 KB, 64-col H chunk
    __shared__ _Float16 W1s[32 * 512];             // 32 KB, staged W1 chunk frags
    __shared__ _Float16 W2s[16 * 512];             // 16 KB, staged W2 chunk frags

    // map block -> (type, tile-within-type)
    int b = blockIdx.x;
    int ty = -1, row0 = 0, valid = 0;
    for (int i = 0; i < NTYPES; ++i) {
        int c = counts[i];
        int nt = (c + TILE_M - 1) >> 7;
        if (b < nt) {
            ty = i;
            row0 = offsets[i] + b * TILE_M;
            valid = c - b * TILE_M;
            if (valid > TILE_M) valid = TILE_M;
            break;
        }
        b -= nt;
    }
    if (ty < 0) return;

    // --- stage A tile (gather rows via perm, f32 -> f16) ---
    {
        int r   = threadIdx.x >> 1;           // row 0..127
        int seg = threadIdx.x & 1;            // 0: from, 1: to
        int e   = perm[row0 + (r < valid ? r : 0)];
        const float4* src = (const float4*)((seg == 0 ? from_e : to_e) + (long)e * DEMB);
        _Float16* dst = &Alds[r * A_STRIDE + seg * DEMB];
#pragma unroll
        for (int j = 0; j < DEMB / 4; ++j) {
            float4 v = src[j];
            v4h h; h.x = (_Float16)v.x; h.y = (_Float16)v.y;
            h.z = (_Float16)v.z; h.w = (_Float16)v.w;
            *(v4h*)(dst + j * 4) = h;
        }
    }

    int lane = threadIdx.x & 31;
    int wave = threadIdx.x >> 5;               // 8 waves * 16 rows = 128 rows
    int mrow = wave * 16;
    int ncol = lane & 15;
    int hi   = lane >> 4;

    v8f acc2[8];
#pragma unroll
    for (int nn = 0; nn < 8; ++nn) acc2[nn] = (v8f)(0.0f);

#pragma unroll 1
    for (int chunk = 0; chunk < MIDC / 64; ++chunk) {     // 8 chunks of 64 MID cols
        // ---- stage this chunk's weight fragments into LDS (once per block) ----
        __syncthreads();   // previous chunk fully consumed (also covers A-tile store)
        stage_weights(w1p + (size_t)chunk * 32 * 512,
                      w2p + (size_t)(ty * 16 + chunk * 2) * 8 * 512,
                      W1s, W2s, threadIdx.x);
        __syncthreads();   // staged weights visible to all waves

        // ---- GEMM1 chunk: H[128 x 64] = relu(A @ W1[:,chunk] + b1) ----
        v8f acc1[4];
#pragma unroll
        for (int n = 0; n < 4; ++n) acc1[n] = (v8f)(0.0f);
#pragma unroll
        for (int kt = 0; kt < K1 / 32; ++kt) {
            v16h a = load_afrag_lds(Alds, mrow, kt * 32, lane, A_STRIDE);
#pragma unroll
            for (int ntl = 0; ntl < 4; ++ntl) {
                v16h bf = load_bfrag_lds(W1s, ntl * 8 + kt, lane);
                acc1[ntl] = __builtin_amdgcn_wmma_f32_16x16x32_f16(
                    false, a, false, bf, (short)0, acc1[ntl], false, false);
            }
        }
        // bias + relu + C-layout -> row-major f16 in LDS (wave-private rows)
#pragma unroll
        for (int ntl = 0; ntl < 4; ++ntl) {
            float bias = b1[chunk * 64 + ntl * 16 + ncol];
#pragma unroll
            for (int r = 0; r < 8; ++r) {
                float v = acc1[ntl][r] + bias;
                v = v > 0.0f ? v : 0.0f;
                int m = mrow + hi * 8 + r;
                Hlds[m * H_STRIDE + ntl * 16 + ncol] = (_Float16)v;
            }
        }
        // ---- GEMM2 partial: acc2 += H_chunk @ W2[chunk, ty*128:+128] ----
#pragma unroll
        for (int kk = 0; kk < 2; ++kk) {
            v16h a2 = load_afrag_lds(Hlds, mrow, kk * 32, lane, H_STRIDE);
#pragma unroll
            for (int nn = 0; nn < 8; ++nn) {
                v16h bf = load_bfrag_lds(W2s, kk * 8 + nn, lane);
                acc2[nn] = __builtin_amdgcn_wmma_f32_16x16x32_f16(
                    false, a2, false, bf, (short)0, acc2[nn], false, false);
            }
        }
    }

    // ---- epilogue: + b2, scatter rows back to out[edge] ----
    float bias2[8];
#pragma unroll
    for (int nn = 0; nn < 8; ++nn) bias2[nn] = b2[ty * OUTC + nn * 16 + ncol];
#pragma unroll
    for (int r = 0; r < 8; ++r) {
        int ml = mrow + hi * 8 + r;
        if (ml < valid) {
            int e = perm[row0 + ml];
            float* dst = out + (long)e * OUTC + ncol;
#pragma unroll
            for (int nn = 0; nn < 8; ++nn)
                dst[nn * 16] = acc2[nn][r] + bias2[nn];
        }
    }
}

// ---------------------------------------------------------------------------
extern "C" void kernel_launch(void* const* d_in, const int* in_sizes, int n_in,
                              void* d_out, int out_size, void* d_ws, size_t ws_size,
                              hipStream_t stream) {
    const float* from_e = (const float*)d_in[0];
    const float* to_e   = (const float*)d_in[1];
    const int*   types  = (const int*)d_in[2];
    const float* W1     = (const float*)d_in[3];
    const float* b1     = (const float*)d_in[4];
    const float* W2     = (const float*)d_in[5];
    const float* b2     = (const float*)d_in[6];
    float* out = (float*)d_out;

    const int nE = in_sizes[2];

    // workspace layout
    int* wsi = (int*)d_ws;
    int* cnts    = wsi;            // 8
    int* offs    = wsi + 8;        // 8
    int* cursors = wsi + 16;       // 8 (+8 pad)
    int* perm    = wsi + 32;       // nE
    size_t half_off = (size_t)(32 + nE) * 4;
    half_off = (half_off + 31) & ~(size_t)31;          // 32B align
    _Float16* w1p = (_Float16*)((char*)d_ws + half_off);               // 256KB
    _Float16* w2p = (_Float16*)((char*)d_ws + half_off + (size_t)K1 * MIDC * 2);

    zero_kernel<<<1, 64, 0, stream>>>(wsi);
    hist_kernel<<<(nE + 255) / 256, 256, 0, stream>>>(types, nE, cnts);
    prefix_kernel<<<1, 32, 0, stream>>>(cnts, offs);
    scatter_kernel<<<(nE + 255) / 256, 256, 0, stream>>>(types, nE, offs, cursors, perm);
    pack_w1_kernel<<<(K1 * MIDC + 255) / 256, 256, 0, stream>>>(W1, w1p);
    pack_w2_kernel<<<(MIDC * OUTC * NTYPES + 255) / 256, 256, 0, stream>>>(W2, w2p);

    int ntiles = (nE + TILE_M - 1) / TILE_M + NTYPES;   // upper bound over all bins
    mlp_tile_kernel<<<ntiles, 256, 0, stream>>>(from_e, to_e, perm, cnts, offs,
                                                w1p, w2p, b1, b2, out);
}
